// G2S_VAE_30107720745238
// MI455X (gfx1250) — compile-verified, matrix operand order checked by reading, not run on hardware
//
#include <hip/hip_runtime.h>
#include <hip/hip_bf16.h>
#include <cstdint>
#include <cstddef>

// ---------------- problem constants ----------------
#define NN   80000      // nodes
#define NE   160000     // edges
#define HE   80000      // half edges
#define HID  300
#define HIDP 304        // 19 * 16
#define KH   320        // hidden K padded to 32
#define NDIM 133
#define EDIM 14
#define NG   1600
#define NTI  19         // N tiles of 16 covering 304
#define NTPAD 20        // padded tile count (guard tile for speculative loads)

typedef __attribute__((ext_vector_type(8)))  float   v8f;
typedef __attribute__((ext_vector_type(16))) __bf16  v16bf;
typedef __attribute__((ext_vector_type(8)))  __bf16  v8bf;

// =====================================================================
// Weight pack: fp32 W[Kact][300] -> bf16 B-fragments.
// Fragment layout (per 16x16x32 tile, 32 lanes x 16 halfs, contiguous):
//   lane l: N = nt*16 + (l&15)
//           K = kt*32 + (l<16 ? (e<8? e : 8+e) : (e<8? 8+e : 16+e))
// =====================================================================
__global__ __launch_bounds__(32)
void kpack(const float* __restrict__ W, int Kact, __bf16* __restrict__ outp, int ktiles)
{
    const int nt = blockIdx.x, kt = blockIdx.y, lane = threadIdx.x;
    const int n  = nt * 16 + (lane & 15);
    const int kb = kt * 32 + ((lane < 16) ? 0 : 8);
    __bf16* dst = outp + (((size_t)nt * ktiles + kt) * 512) + (size_t)lane * 16;
#pragma unroll
    for (int e = 0; e < 16; ++e) {
        const int k = kb + ((e < 8) ? e : (8 + e));   // e>=8 -> 16 + (e-8)
        float v = (k < Kact && n < HID) ? W[(size_t)k * HID + n] : 0.f;
        dst[e] = (__bf16)v;
    }
}

// =====================================================================
// A builders (fused gather / concat / subtract, fp32 -> bf16, zero pad)
// =====================================================================
__global__ void kbuild_lin(const float* __restrict__ nf, const __bf16* __restrict__ nb,
                           int D,
                           const float* __restrict__ ea,
                           const int* __restrict__ sh, const int* __restrict__ dh,
                           __bf16* __restrict__ Aout, int Kpad)
{
    const int e = blockIdx.x, k = threadIdx.x;          // blockDim == Kpad
    const int s = (e < HE) ? sh[e] : dh[e - HE];        // src = [src_half, dst_half]
    float v = 0.f;
    if (k < D)              v = nf ? nf[(size_t)s * D + k] : (float)nb[(size_t)s * KH + k];
    else if (k < D + EDIM)  v = ea[(size_t)e * EDIM + (k - D)];
    Aout[(size_t)e * Kpad + k] = (__bf16)v;
}

__global__ void kbuild_mp(const float* __restrict__ inc, const __bf16* __restrict__ h,
                          const int* __restrict__ sh, const int* __restrict__ dh,
                          __bf16* __restrict__ Aout)
{
    const int e = blockIdx.x, k = threadIdx.x;          // blockDim == KH
    const int s = (e < HE) ? sh[e] : dh[e - HE];
    const int r = (e < HE) ? (e + HE) : (e - HE);       // rev edge
    float v = 0.f;
    if (k < HID) v = inc[(size_t)s * HIDP + k] - (float)h[(size_t)r * KH + k];
    Aout[(size_t)e * KH + k] = (__bf16)v;
}

__global__ void kbuild_au(const float* __restrict__ nf, const __bf16* __restrict__ nb,
                          int D, const float* __restrict__ inc,
                          __bf16* __restrict__ Aout, int Kpad)
{
    const int n = blockIdx.x, k = threadIdx.x;          // blockDim == Kpad
    float v = 0.f;
    if (k < D)             v = nf ? nf[(size_t)n * D + k] : (float)nb[(size_t)n * KH + k];
    else if (k < D + HID)  v = inc[(size_t)n * HIDP + (k - D)];
    Aout[(size_t)n * Kpad + k] = (__bf16)v;
}

// segment_sum over incoming edges: inc[dst[e]] += h[e]
__global__ void kscatter(const __bf16* __restrict__ h,
                         const int* __restrict__ sh, const int* __restrict__ dh,
                         float* __restrict__ inc)
{
    const int e = blockIdx.x, k = threadIdx.x;          // blockDim == KH
    if (k >= HID) return;
    const int d = (e < HE) ? dh[e] : sh[e - HE];        // dst = [dst_half, src_half]
    atomicAdd(&inc[(size_t)d * HIDP + k], (float)h[(size_t)e * KH + k]);
}

// =====================================================================
// LDS-staged WMMA GEMM.
// Block = 320 threads (10 waves).  Block computes a 32-row x 304-col
// output panel: the contiguous 32 x Kpad bf16 A panel is staged into
// LDS once (ds path), then wave w computes n-tiles {2w, 2w+1} x
// m-subtiles {0,1} -> 4 accumulators, reusing each A fragment across
// 2 WMMAs and each B fragment across 2 WMMAs.
// flags: 1=relu, 2=+resid, 4=pool-atomic.
// =====================================================================
__global__ __launch_bounds__(320)
void gemm_wmma(const __bf16* __restrict__ A, int lda,
               const __bf16* __restrict__ Bp, int ktiles,
               const float* __restrict__ bias,
               const __bf16* __restrict__ resid, int ldr,
               __bf16* __restrict__ outb, int ldo,
               const int* __restrict__ batch,
               const float* __restrict__ wat,
               float* __restrict__ pool,
               int flags)
{
    extern __shared__ char smem_raw[];
    __bf16* smem = (__bf16*)smem_raw;

    const int mt = blockIdx.x;                 // 32-row panel index

    // ---- stage A panel (contiguous: lda == Kpad) into LDS ----
    {
        const uint4* __restrict__ src = (const uint4*)(A + (size_t)mt * 32 * lda);
        uint4* dst = (uint4*)smem;
        const int nchunk = (32 * lda) >> 3;    // 16B chunks
        for (int c = threadIdx.x; c < nchunk; c += blockDim.x) dst[c] = src[c];
    }
    __syncthreads();

    const int wave = threadIdx.x >> 5;
    const int lane = threadIdx.x & 31;
    const int rl   = lane & 15;
    const int ksel = (lane < 16) ? 0 : 8;
    const int nt0  = wave * 2;
    const int nt1  = wave * 2 + 1;             // may be 19 (guard tile)

    const __bf16* __restrict__ B0 = Bp + ((size_t)nt0 * ktiles) * 512 + (size_t)lane * 16;
    const __bf16* __restrict__ B1 = Bp + ((size_t)nt1 * ktiles) * 512 + (size_t)lane * 16;
    const __bf16* a0p = smem + (size_t)rl * lda + ksel;        // m-subtile 0
    const __bf16* a1p = a0p + (size_t)16 * lda;                // m-subtile 1

    v8f acc00 = {0.f,0.f,0.f,0.f,0.f,0.f,0.f,0.f};
    v8f acc01 = acc00, acc10 = acc00, acc11 = acc00;

    for (int kt = 0; kt < ktiles; ++kt) {
        v8bf a0lo = *(const v8bf*)(a0p);
        v8bf a0hi = *(const v8bf*)(a0p + 16);
        v8bf a1lo = *(const v8bf*)(a1p);
        v8bf a1hi = *(const v8bf*)(a1p + 16);
        a0p += 32; a1p += 32;
        v16bf af0, af1;
#pragma unroll
        for (int i = 0; i < 8; ++i) {
            af0[i] = a0lo[i]; af0[i + 8] = a0hi[i];
            af1[i] = a1lo[i]; af1[i + 8] = a1hi[i];
        }
        v16bf bf0 = *(const v16bf*)(B0);
        v16bf bf1 = *(const v16bf*)(B1);
        B0 += 512; B1 += 512;
        __builtin_prefetch(B0, 0, 1);          // global_prefetch_b8
        acc00 = __builtin_amdgcn_wmma_f32_16x16x32_bf16(false, af0, false, bf0, (short)0, acc00, false, false);
        acc01 = __builtin_amdgcn_wmma_f32_16x16x32_bf16(false, af0, false, bf1, (short)0, acc01, false, false);
        acc10 = __builtin_amdgcn_wmma_f32_16x16x32_bf16(false, af1, false, bf0, (short)0, acc10, false, false);
        acc11 = __builtin_amdgcn_wmma_f32_16x16x32_bf16(false, af1, false, bf1, (short)0, acc11, false, false);
    }

    // ---- epilogue (per accumulator tile) ----
    const int hi8 = (lane >= 16) ? 8 : 0;
    auto epi = [&](const v8f& acc, int nt, int msub) {
        const int col = nt * 16 + rl;
        if (nt >= NTI || col >= HID) return;
        const float b = bias[col];
#pragma unroll
        for (int i = 0; i < 8; ++i) {
            const int r = mt * 32 + msub * 16 + i + hi8;   // C layout: VGPR i -> M=i / 8+i
            float v = acc[i] + b;
            if (flags & 2) v += (float)resid[(size_t)r * ldr + col];
            if (flags & 1) v = v > 0.f ? v : 0.f;
            if (flags & 4) {
                atomicAdd(&pool[(size_t)batch[r] * HID + col], v * wat[r]);
            } else {
                outb[(size_t)r * ldo + col] = (__bf16)v;
            }
        }
    };
    epi(acc00, nt0, 0);
    epi(acc01, nt1, 0);
    epi(acc10, nt0, 1);
    epi(acc11, nt1, 1);
}

// =====================================================================
// misc small kernels
// =====================================================================
__global__ void kfill(float* __restrict__ p, long n)
{
    long i = (long)blockIdx.x * blockDim.x + threadIdx.x;
    if (i < n) p[i] = 0.f;
}

__global__ void kcount(const int* __restrict__ b, float* __restrict__ c)
{
    int n = blockIdx.x * blockDim.x + threadIdx.x;
    if (n < NN) atomicAdd(&c[b[n]], 1.f);
}

__global__ void kfinal(const float* __restrict__ mug, const float* __restrict__ lvg,
                       const float* __restrict__ cnt, const float* __restrict__ eps,
                       float* __restrict__ out)
{
    int i = blockIdx.x * blockDim.x + threadIdx.x;
    if (i >= NG * HID) return;
    int g = i / HID;
    float cm = cnt[g]; cm = cm > 1.f ? cm : 1.f;
    float m = mug[i] / cm;
    float l = lvg[i] / cm;
    out[i] = m + __expf(0.5f * l) * eps[i];
}

// =====================================================================
// host orchestration
// =====================================================================
extern "C" void kernel_launch(void* const* d_in, const int* in_sizes, int n_in,
                              void* d_out, int out_size, void* d_ws, size_t ws_size,
                              hipStream_t stream)
{
    (void)in_sizes; (void)n_in; (void)out_size; (void)ws_size;

    const float* x        = (const float*)d_in[0];
    const float* ea       = (const float*)d_in[1];
    const float* wat      = (const float*)d_in[2];
    const float* eps      = (const float*)d_in[3];
    const int*   sh       = (const int*)  d_in[4];
    const int*   dh       = (const int*)  d_in[5];
    const int*   batch    = (const int*)  d_in[6];
    const float* t_lin_w  = (const float*)d_in[7];
    const float* t_lin_b  = (const float*)d_in[8];
    const float* t_mp_w   = (const float*)d_in[9];
    const float* t_mp_b   = (const float*)d_in[10];
    const float* t_au_w   = (const float*)d_in[11];
    const float* t_au_b   = (const float*)d_in[12];
    const float* mu_lin_w = (const float*)d_in[13];
    const float* mu_lin_b = (const float*)d_in[14];
    const float* mu_mp_w  = (const float*)d_in[15];
    const float* mu_mp_b  = (const float*)d_in[16];
    const float* mu_au_w  = (const float*)d_in[17];
    const float* mu_au_b  = (const float*)d_in[18];
    const float* lv_lin_w = (const float*)d_in[19];
    const float* lv_lin_b = (const float*)d_in[20];
    const float* lv_mp_w  = (const float*)d_in[21];
    const float* lv_mp_b  = (const float*)d_in[22];
    const float* lv_au_w  = (const float*)d_in[23];
    const float* lv_au_b  = (const float*)d_in[24];

    // ---- workspace carve-out ----
    char* base = (char*)d_ws;
    size_t off = 0;
    auto alloc = [&](size_t bytes) -> char* {
        size_t o = off;
        off = (off + bytes + 255) & ~(size_t)255;
        return base + o;
    };
    __bf16* Hb  = (__bf16*)alloc((size_t)NE * KH * 2);     // h activations (bf16, K-padded)
    __bf16* Ab  = (__bf16*)alloc((size_t)NE * KH * 2);     // A staging (also >= NN*608*2)
    __bf16* SHb = (__bf16*)alloc((size_t)NN * KH * 2);     // shared node features
    float*  inc = (float*) alloc((size_t)NN * HIDP * 4);   // segment-sum accumulator
    float*  mug = (float*) alloc((size_t)NG * HID * 4);
    float*  lvg = (float*) alloc((size_t)NG * HID * 4);
    float*  cnt = (float*) alloc((size_t)NG * 4);
    auto wpack = [&](int Kpad) -> __bf16* {
        // NTPAD tiles: one extra guard n-tile so speculative nt==19 loads stay in-bounds
        return (__bf16*)alloc((size_t)(Kpad / 32) * NTPAD * 512 * 2);
    };
    __bf16* pTlin = wpack(160);
    __bf16* pTmp[3] = { wpack(320), wpack(320), wpack(320) };
    __bf16* pTau  = wpack(448);
    __bf16* pMlin = wpack(320);
    __bf16* pMmp[3] = { wpack(320), wpack(320), wpack(320) };
    __bf16* pMau  = wpack(608);
    __bf16* pLlin = wpack(320);
    __bf16* pLmp[3] = { wpack(320), wpack(320), wpack(320) };
    __bf16* pLau  = wpack(608);

    // ---- pack all weights to bf16 WMMA fragments ----
    auto pack = [&](const float* W, int Kact, int Kpad, __bf16* dst) {
        kpack<<<dim3(NTI, Kpad / 32), 32, 0, stream>>>(W, Kact, dst, Kpad / 32);
    };
    pack(t_lin_w, NDIM + EDIM, 160, pTlin);
    for (int i = 0; i < 3; ++i) pack(t_mp_w  + (size_t)i * HID * HID, HID, 320, pTmp[i]);
    pack(t_au_w, NDIM + HID, 448, pTau);
    pack(mu_lin_w, HID + EDIM, 320, pMlin);
    for (int i = 0; i < 3; ++i) pack(mu_mp_w + (size_t)i * HID * HID, HID, 320, pMmp[i]);
    pack(mu_au_w, 2 * HID, 608, pMau);
    pack(lv_lin_w, HID + EDIM, 320, pLlin);
    for (int i = 0; i < 3; ++i) pack(lv_mp_w + (size_t)i * HID * HID, HID, 320, pLmp[i]);
    pack(lv_au_w, 2 * HID, 608, pLau);

    auto zero = [&](float* p, size_t n) {
        kfill<<<(unsigned)((n + 255) / 256), 256, 0, stream>>>(p, (long)n);
    };

    // ---- pooling state ----
    zero(mug, (size_t)NG * HID);
    zero(lvg, (size_t)NG * HID);
    zero(cnt, (size_t)NG);
    kcount<<<(NN + 255) / 256, 256, 0, stream>>>(batch, cnt);

    auto gemm = [&](const __bf16* Am, int Mrows, int Kpad, const __bf16* Bpk,
                    const float* bias, const __bf16* res, int ldr,
                    __bf16* ob, int ldo, const int* bt, const float* wa, float* pl, int flags) {
        gemm_wmma<<<dim3(Mrows / 32), 320, (size_t)32 * Kpad * 2, stream>>>(
            Am, Kpad, Bpk, Kpad / 32, bias, res, ldr, ob, ldo, bt, wa, pl, flags);
    };

    // ---- one conv pass ----
    auto conv = [&](const float* nf, const __bf16* nb, int D, int KlinP,
                    __bf16* pLin, const float* linB,
                    __bf16* const* pMp, const float* mpB,
                    __bf16* pAu, const float* auB, int KauP,
                    __bf16* outSh, float* poolOut)
    {
        // h = relu(concat(node_in[src], edge_attr) @ lin_w + lin_b)
        kbuild_lin<<<NE, KlinP, 0, stream>>>(nf, nb, D, ea, sh, dh, Ab, KlinP);
        gemm(Ab, NE, KlinP, pLin, linB, nullptr, 0, Hb, KH,
             nullptr, nullptr, nullptr, /*relu*/1);

        // 3x: h = relu(h + (inc[src] - h[rev]) @ mp_w[i] + mp_b[i])
        for (int i = 0; i < 3; ++i) {
            zero(inc, (size_t)NN * HIDP);
            kscatter <<<NE, KH, 0, stream>>>(Hb, sh, dh, inc);
            kbuild_mp<<<NE, KH, 0, stream>>>(inc, Hb, sh, dh, Ab);
            gemm(Ab, NE, KH, pMp[i], mpB + (size_t)i * HID, Hb, KH, Hb, KH,
                 nullptr, nullptr, nullptr, /*relu|resid*/3);
        }

        // a = concat(node_in, segment_sum(h, dst)) @ au_w + au_b
        zero(inc, (size_t)NN * HIDP);
        kscatter<<<NE, KH, 0, stream>>>(Hb, sh, dh, inc);
        kbuild_au<<<NN, KauP, 0, stream>>>(nf, nb, D, inc, Ab, KauP);
        if (outSh) {
            gemm(Ab, NN, KauP, pAu, auB, nullptr, 0, outSh, KH,
                 nullptr, nullptr, nullptr, /*relu*/1);
        } else {
            // fused weighted graph pooling (no relu)
            gemm(Ab, NN, KauP, pAu, auB, nullptr, 0, nullptr, 0,
                 batch, wat, poolOut, /*pool*/4);
        }
    };

    // shared = conv(x, ..., relu)
    conv(x, nullptr, NDIM, 160, pTlin, t_lin_b, pTmp, t_mp_b, pTau, t_au_b, 448, SHb, nullptr);
    // mu / lv convs with fused pooling
    conv(nullptr, SHb, HID, 320, pMlin, mu_lin_b, pMmp, mu_mp_b, pMau, mu_au_b, 608, nullptr, mug);
    conv(nullptr, SHb, HID, 320, pLlin, lv_lin_b, pLmp, lv_mp_b, pLau, lv_au_b, 608, nullptr, lvg);

    // z = mu_g/cnt + exp(0.5*lv_g/cnt) * eps
    kfinal<<<(NG * HID + 255) / 256, 256, 0, stream>>>(mug, lvg, cnt, eps, (float*)d_out);
}
